// SurfNNODE_59004260713080
// MI455X (gfx1250) — compile-verified, matrix operand order checked by reading, not run on hardware
//
#include <hip/hip_runtime.h>
#include <hip/hip_bf16.h>

typedef __attribute__((ext_vector_type(16))) _Float16 v16h;
typedef __attribute__((ext_vector_type(8)))  _Float16 v8h;
typedef __attribute__((ext_vector_type(8)))  float    v8f;

// ---------------------------------------------------------------------------
// Sizes (fixed by the reference)
// ---------------------------------------------------------------------------
#define NCH      32          // C2
#define KCONV    864         // 32*27
#define VOLSTEP  (64*64*64)  // per-channel stride = 262144

// workspace layout (halfs) — weights stored in WMMA-B-fragment order:
//   index = ((nt*KB + kb)*32 + lane)*16 + h
#define WC_OFF   0
#define WC_SZ    (864*64)               // 55296
#define WL_OFF   (WC_OFF+WC_SZ)
#define WL_SZ    (64*128)               // 8192
#define W2_OFF   (WL_OFF+WL_SZ)
#define W2_SZ    (256*512)              // 131072
#define W3_OFF   (W2_OFF+W2_SZ)
#define W3_SZ    (512*256)              // 131072
#define WTOTAL   (W3_OFF+W3_SZ)        // 325632 halfs = 651,264 B

// ---------------------------------------------------------------------------
// Weight swizzle: f32 [N_out, K_in] row-major -> f16 fragment-ready layout.
// Fragment half (lane l, half h) holds B[k][n] with
//   k = kb*32 + ((h&8)<<1) + ((l>=16)?8:0) + (h&7),  n = nt*16 + (l&15)
// (per ISA 7.12.2 16-bit operand VGPR layout, B symmetric to A over K).
// ---------------------------------------------------------------------------
__device__ __forceinline__ void swizzle_one(const float* __restrict__ W,
                                            _Float16* __restrict__ dst,
                                            int K, int KB, int i) {
    const int h  = i & 15;
    const int l  = (i >> 4) & 31;
    const int f  = i >> 9;           // fragment = nt*KB + kb
    const int kb = f % KB;
    const int nt = f / KB;
    const int k  = kb * 32 + ((h & 8) << 1) + ((l >> 4) << 3) + (h & 7);
    const int n  = nt * 16 + (l & 15);
    dst[i] = (_Float16)W[n * K + k];
}

__global__ void convert_weights(const float* __restrict__ conv_w,
                                const float* __restrict__ lfc_w,
                                const float* __restrict__ fc2_w,
                                const float* __restrict__ fc3_w,
                                _Float16* __restrict__ ws) {
    int i = blockIdx.x * blockDim.x + threadIdx.x;
    if (i < WC_SZ) { swizzle_one(conv_w, ws + WC_OFF, 864,  27, i); return; }
    i -= WC_SZ;
    if (i < WL_SZ) { swizzle_one(lfc_w,  ws + WL_OFF, 64,   2,  i); return; }
    i -= WL_SZ;
    if (i < W2_SZ) { swizzle_one(fc2_w,  ws + W2_OFF, 256,  8,  i); return; }
    i -= W2_SZ;
    if (i < W3_SZ) { swizzle_one(fc3_w,  ws + W3_OFF, 512,  16, i); return; }
}

// ---------------------------------------------------------------------------
// WMMA tile GEMM: A from LDS (row-major f16), B from fragment-ready global.
// ---------------------------------------------------------------------------
__device__ __forceinline__ v8f wmma_gemm(const _Float16* __restrict__ aTile, int astride,
                                         const _Float16* __restrict__ wSwz, int KB,
                                         int lane, int nt) {
    v8f acc = {};
    const int row = lane & 15;
    const int kl  = (lane >> 4) << 3;
    const _Float16* ap = aTile + row * astride + kl;           // 16B aligned
    const _Float16* bp = wSwz + (nt * KB) * 512 + lane * 16;   // 32B aligned
    for (int kb = 0; kb < KB; ++kb) {
        const v16h b  = *(const v16h*)bp;                      // 2x b128, coalesced
        const v8h alo = *(const v8h*)(ap + kb * 32);           // ds_load_b128
        const v8h ahi = *(const v8h*)(ap + kb * 32 + 16);      // ds_load_b128
        v16h a;
        #pragma unroll
        for (int h = 0; h < 8; ++h) { a[h] = alo[h]; a[h + 8] = ahi[h]; }
        acc = __builtin_amdgcn_wmma_f32_16x16x32_f16(
                  false, a, false, b, (short)0, acc, false, false);
        bp += 512;
    }
    return acc;
}

// C/D layout per ISA: row = e + (lane>=16?8:0), col = lane&15
__device__ __forceinline__ void store_tile(_Float16* __restrict__ tile, int stride,
                                           int lane, int n0, v8f c,
                                           const float* __restrict__ bias, bool act) {
    const int col   = lane & 15;
    const int rbase = (lane >> 4) << 3;
    const float bv  = bias[n0 + col];
    #pragma unroll
    for (int e = 0; e < 8; ++e) {
        float v = c[e] + bv;
        if (act) v = (v >= 0.0f) ? v : 0.2f * v;
        tile[(rbase + e) * stride + n0 + col] = (_Float16)v;
    }
}

__device__ __forceinline__ float grid_val(int a) {
    // np.linspace(-3//2, 3//2, 3) = [-2, -0.5, 1]
    return (a == 0) ? -2.0f : ((a == 1) ? -0.5f : 1.0f);
}

// ---------------------------------------------------------------------------
// Fused per-16-point pipeline (1 workgroup = 4 waves = 16 points)
// ---------------------------------------------------------------------------
__global__ __launch_bounds__(128) void surfnn_fused(
    const float* __restrict__ x,                       // [m,3]
    const float* __restrict__ vol0, const float* __restrict__ vol1,
    const float* __restrict__ vol2, const float* __restrict__ vol3,
    const _Float16* __restrict__ wc_s, const float* __restrict__ conv_b,
    const _Float16* __restrict__ wl_s, const float* __restrict__ lfc_b,
    const float* __restrict__ fc1_w,   const float* __restrict__ fc1_b,
    const _Float16* __restrict__ w2_s, const float* __restrict__ fc2_b,
    const _Float16* __restrict__ w3_s, const float* __restrict__ fc3_b,
    const float* __restrict__ fc4_w,   const float* __restrict__ fc4_b,
    float* __restrict__ out, int m)
{
    const int tid  = threadIdx.x;
    const int lane = tid & 31;
    const int wave = tid >> 5;
    const int p0   = blockIdx.x * 16;

    __shared__ __align__(32) _Float16 sCube[16 * KCONV];  // 27648 B
    __shared__ __align__(32) _Float16 sZ64 [16 * 64];     //  2048 B
    __shared__ __align__(32) _Float16 sZ   [16 * 256];    //  8192 B (point|local)
    __shared__ __align__(32) _Float16 sH2  [16 * 512];    // 16384 B
    __shared__ __align__(32) _Float16 sH3  [16 * 256];    //  8192 B

    const float* vols[4] = {vol0, vol1, vol2, vol3};

    // ---- Phase 0a: point branch  z_point = leaky(pts @ fc1_w.T + b) -------
    {
        const int p  = tid & 15;
        const int ob = (tid >> 4) * 16;
        const int gp = (p0 + p < m) ? (p0 + p) : (m - 1);
        const float px = x[gp * 3 + 0], py = x[gp * 3 + 1], pz = x[gp * 3 + 2];
        #pragma unroll
        for (int oo = 0; oo < 16; ++oo) {
            const int o = ob + oo;
            float v = fc1_w[o * 3 + 0] * px + fc1_w[o * 3 + 1] * py
                    + fc1_w[o * 3 + 2] * pz + fc1_b[o];
            v = (v >= 0.0f) ? v : 0.2f * v;
            sZ[p * 256 + o] = (_Float16)v;
        }
    }

    // ---- Phase 0b: multiscale trilinear sampling -> cube features ---------
    for (int task = tid; task < 16 * 27; task += 128) {
        const int r  = task / 27;
        const int j  = task - r * 27;
        const int gp = (p0 + r < m) ? (p0 + r) : (m - 1);
        const float px = x[gp * 3 + 0], py = x[gp * 3 + 1], pz = x[gp * 3 + 2];
        const int ja = j / 9, jb = (j / 3) % 3, jc = j % 3;
        const float gx = grid_val(ja), gy = grid_val(jb), gz = grid_val(jc);

        float acc[NCH];
        #pragma unroll
        for (int c = 0; c < NCH; ++c) acc[c] = 0.0f;

        for (int q = 0; q < 4; ++q) {
            const float s  = (2.0f * (float)(1 << q)) / 64.0f;   // 2*2^q/Dm
            const float cx = px + gx * s;                         // rescale == 1
            const float cy = py + gy * s;
            const float cz = pz + gz * s;
            // align_corners=True, border padding
            const float ix = fminf(fmaxf((cx + 1.0f) * 0.5f * 63.0f, 0.0f), 63.0f);
            const float iy = fminf(fmaxf((cy + 1.0f) * 0.5f * 63.0f, 0.0f), 63.0f);
            const float iz = fminf(fmaxf((cz + 1.0f) * 0.5f * 63.0f, 0.0f), 63.0f);
            const float x0f = floorf(ix), y0f = floorf(iy), z0f = floorf(iz);
            const int x0 = (int)x0f, y0 = (int)y0f, z0 = (int)z0f;
            const int x1 = (x0 + 1 < 63) ? x0 + 1 : 63;
            const int y1 = (y0 + 1 < 63) ? y0 + 1 : 63;
            const int z1 = (z0 + 1 < 63) ? z0 + 1 : 63;
            const float fx = ix - x0f, fy = iy - y0f, fz = iz - z0f;
            const float w000 = (1-fz)*(1-fy)*(1-fx), w001 = (1-fz)*(1-fy)*fx;
            const float w010 = (1-fz)*fy*(1-fx),     w011 = (1-fz)*fy*fx;
            const float w100 = fz*(1-fy)*(1-fx),     w101 = fz*(1-fy)*fx;
            const float w110 = fz*fy*(1-fx),         w111 = fz*fy*fx;
            const int i000 = (z0*64 + y0)*64 + x0, i001 = (z0*64 + y0)*64 + x1;
            const int i010 = (z0*64 + y1)*64 + x0, i011 = (z0*64 + y1)*64 + x1;
            const int i100 = (z1*64 + y0)*64 + x0, i101 = (z1*64 + y0)*64 + x1;
            const int i110 = (z1*64 + y1)*64 + x0, i111 = (z1*64 + y1)*64 + x1;
            const float* vq = vols[q];
            #pragma unroll
            for (int c = 0; c < NCH; ++c) {
                const float* vc = vq + c * VOLSTEP;
                acc[c] += vc[i000]*w000 + vc[i001]*w001 + vc[i010]*w010 + vc[i011]*w011
                        + vc[i100]*w100 + vc[i101]*w101 + vc[i110]*w110 + vc[i111]*w111;
            }
        }
        #pragma unroll
        for (int c = 0; c < NCH; ++c)
            sCube[r * KCONV + c * 27 + j] = (_Float16)acc[c];   // feature = c*27+j
    }
    __syncthreads();

    // ---- Phase 1: conv GEMM  [16,864] x [864,64] --------------------------
    {
        const int nt = wave;                            // 4 waves, 4 tiles
        v8f acc = wmma_gemm(sCube, KCONV, wc_s, 27, lane, nt);
        store_tile(sZ64, 64, lane, nt * 16, acc, conv_b, false);
    }
    __syncthreads();

    // ---- Phase 2: lfc GEMM  [16,64] x [64,128] -> z_local -----------------
    for (int nt = wave; nt < 8; nt += 4) {
        v8f acc = wmma_gemm(sZ64, 64, wl_s, 2, lane, nt);
        store_tile(sZ + 128, 256, lane, nt * 16, acc, lfc_b, false); // cols 128..255
    }
    __syncthreads();

    // ---- Phase 3: fc2  [16,256] x [256,512], leaky ------------------------
    for (int nt = wave; nt < 32; nt += 4) {
        v8f acc = wmma_gemm(sZ, 256, w2_s, 8, lane, nt);
        store_tile(sH2, 512, lane, nt * 16, acc, fc2_b, true);
    }
    __syncthreads();

    // ---- Phase 4: fc3  [16,512] x [512,256], leaky ------------------------
    for (int nt = wave; nt < 16; nt += 4) {
        v8f acc = wmma_gemm(sH2, 512, w3_s, 16, lane, nt);
        store_tile(sH3, 256, lane, nt * 16, acc, fc3_b, true);
    }
    __syncthreads();

    // ---- Phase 5: fc4  [16,256] x [256,3] (f32) ---------------------------
    if (tid < 48) {
        const int p = tid / 3, o = tid % 3;
        const int gp = p0 + p;
        if (gp < m) {
            float acc = fc4_b[o];
            for (int k = 0; k < 256; ++k)
                acc += (float)sH3[p * 256 + k] * fc4_w[o * 256 + k];
            out[gp * 3 + o] = acc;
        }
    }
}

// ---------------------------------------------------------------------------
// Launch
// ---------------------------------------------------------------------------
extern "C" void kernel_launch(void* const* d_in, const int* in_sizes, int n_in,
                              void* d_out, int out_size, void* d_ws, size_t ws_size,
                              hipStream_t stream) {
    const int m = in_sizes[1] / 3;

    const float* x      = (const float*)d_in[1];
    const float* vol0   = (const float*)d_in[2];
    const float* vol1   = (const float*)d_in[3];
    const float* vol2   = (const float*)d_in[4];
    const float* vol3   = (const float*)d_in[5];
    const float* conv_w = (const float*)d_in[6];
    const float* conv_b = (const float*)d_in[7];
    const float* lfc_w  = (const float*)d_in[8];
    const float* lfc_b  = (const float*)d_in[9];
    const float* fc1_w  = (const float*)d_in[10];
    const float* fc1_b  = (const float*)d_in[11];
    const float* fc2_w  = (const float*)d_in[12];
    const float* fc2_b  = (const float*)d_in[13];
    const float* fc3_w  = (const float*)d_in[14];
    const float* fc3_b  = (const float*)d_in[15];
    const float* fc4_w  = (const float*)d_in[16];
    const float* fc4_b  = (const float*)d_in[17];

    _Float16* ws   = (_Float16*)d_ws;
    _Float16* wc_s = ws + WC_OFF;
    _Float16* wl_s = ws + WL_OFF;
    _Float16* w2_s = ws + W2_OFF;
    _Float16* w3_s = ws + W3_OFF;

    convert_weights<<<(WTOTAL + 255) / 256, 256, 0, stream>>>(
        conv_w, lfc_w, fc2_w, fc3_w, ws);

    surfnn_fused<<<(m + 15) / 16, 128, 0, stream>>>(
        x, vol0, vol1, vol2, vol3,
        wc_s, conv_b, wl_s, lfc_b,
        fc1_w, fc1_b, w2_s, fc2_b, w3_s, fc3_b,
        fc4_w, fc4_b,
        (float*)d_out, m);
}
